// ParallelSwinLayer_15848429322945
// MI455X (gfx1250) — compile-verified, matrix operand order checked by reading, not run on hardware
//
#include <hip/hip_runtime.h>

// ---------------------------------------------------------------------------
// ParallelSwinLayer for MI455X (gfx1250, wave32, WMMA bf16 16x16x32)
// Full-sequence attention (window permute cancels), t/dt unused.
// B=2, N=2048, D=1024, H=16, HD=64, MLP=4096.
// ---------------------------------------------------------------------------

#define BB 2
#define NN 2048
#define DD 1024
#define HH 16
#define HD 64
#define MLP_ 4096
#define MTOK (BB * NN)              // 4096 token rows
#define EPSF 1.1920928955078125e-07f

typedef __attribute__((ext_vector_type(16))) __bf16 v16bf;
typedef __attribute__((ext_vector_type(8)))  __bf16 v8bf;
typedef __attribute__((ext_vector_type(8)))  float  v8f;

static __device__ inline v16bf combine8(v8bf a, v8bf b) {
  return __builtin_shufflevector(a, b, 0, 1, 2, 3, 4, 5, 6, 7,
                                       8, 9, 10, 11, 12, 13, 14, 15);
}

static __device__ inline v8f wmma_bf16(v16bf a, v16bf b, v8f c) {
  // D = A(16x32 bf16) x B(32x16 bf16) + C(16x16 f32)
  return __builtin_amdgcn_wmma_f32_16x16x32_bf16(
      false, a, false, b, (short)0, c, false, false);
}

// quad-lane xor via DPP quad_perm (pure VALU, no DS round-trip);
// CTRL must be a compile-time constant for the builtin.
template <int CTRL>
static __device__ inline float dpp_qperm_f(float v) {
  int r = __builtin_amdgcn_update_dpp(0, __float_as_int(v), CTRL, 0xF, 0xF, true);
  return __int_as_float(r);
}

// ---------------------------------------------------------------------------
// f32 -> bf16 cast (weights)
// ---------------------------------------------------------------------------
__global__ __launch_bounds__(256) void cast_bf16_kernel(
    const float* __restrict__ src, __bf16* __restrict__ dst, int n) {
  int i = blockIdx.x * 256 + threadIdx.x;
  if (i < n) dst[i] = (__bf16)src[i];
}

// ---------------------------------------------------------------------------
// RMSNorm: out_bf16[row, :] = x * rsqrt(mean(x^2) + eps) * g
// grid = rows, block = 256
// ---------------------------------------------------------------------------
__global__ __launch_bounds__(256) void rmsnorm_kernel(
    const float* __restrict__ x, const float* __restrict__ g,
    __bf16* __restrict__ out) {
  __shared__ float red[8];
  const int tid = threadIdx.x;
  const int lane = tid & 31, wid = tid >> 5;
  const size_t base = (size_t)blockIdx.x * DD;
  float ss = 0.f;
  for (int i = tid; i < DD; i += 256) {
    float v = x[base + i];
    ss += v * v;
  }
  for (int off = 16; off > 0; off >>= 1) ss += __shfl_xor(ss, off, 32);
  if (lane == 0) red[wid] = ss;
  __syncthreads();
  if (tid == 0) {
    float t = 0.f;
    for (int i = 0; i < 8; ++i) t += red[i];
    red[0] = rsqrtf(t * (1.0f / DD) + EPSF);
  }
  __syncthreads();
  const float r = red[0];
  for (int i = tid; i < DD; i += 256)
    out[base + i] = (__bf16)(x[base + i] * r * g[i]);
}

// ---------------------------------------------------------------------------
// Generic bf16 WMMA GEMM:  C[M, Nout] = A[M, K] @ W[Nout, K]^T  (+ residual)
// Block tile 128(M) x 128(N), BK = 32, 256 threads = 8 waves (4x2), each wave
// 32x64 = 2x4 WMMA tiles. LDS staged via global_load_async_to_lds_b128 with
// double buffering on ASYNCcnt. EPI: 1 = store bf16, 2 = +f32 residual, f32.
// All dims are multiples of the tile sizes for this layer.
// ---------------------------------------------------------------------------
template <int EPI>
__global__ __launch_bounds__(256) void gemm_bf16_kernel(
    const __bf16* __restrict__ A, const __bf16* __restrict__ W,
    int K, int Nout, const float* __restrict__ res,
    float* __restrict__ outF, __bf16* __restrict__ outB) {
  __shared__ __bf16 As[2][128 * 40];   // stride 40 halves (80B, 16B-aligned)
  __shared__ __bf16 Bs[2][128 * 40];
  const int tid = threadIdx.x;
  const int lane = tid & 31, wid = tid >> 5;
  const int waveM = wid >> 1, waveN = wid & 1;   // 4 x 2 wave grid
  const int m0 = blockIdx.y * 128;
  const int n0 = blockIdx.x * 128;
  const int lr = lane & 15;                      // M/N index within 16-tile
  const int g0 = (lane >> 4) * 8;                // K sub-group base
  const int srow = tid >> 2, sc = (tid & 3) * 8; // staging map (2 rows/thread)

  v8f acc[2][4] = {};
  const int nk = K >> 5;

  // async-stage one 128x32 A tile + 128x32 B tile into LDS buffer `buf`
  auto stage = [&](int buf, int k0) {
#pragma unroll
    for (int i = 0; i < 2; ++i) {
      int row = srow + i * 64;
      unsigned la = (unsigned)(size_t)&As[buf][row * 40 + sc];
      unsigned long long ga =
          (unsigned long long)(size_t)&A[(size_t)(m0 + row) * K + k0 + sc];
      asm volatile("global_load_async_to_lds_b128 %0, %1, off"
                   :: "v"(la), "v"(ga) : "memory");
      unsigned lb = (unsigned)(size_t)&Bs[buf][row * 40 + sc];
      unsigned long long gb =
          (unsigned long long)(size_t)&W[(size_t)(n0 + row) * K + k0 + sc];
      asm volatile("global_load_async_to_lds_b128 %0, %1, off"
                   :: "v"(lb), "v"(gb) : "memory");
    }
  };

  stage(0, 0);
  for (int ks = 0; ks < nk; ++ks) {
    const int buf = ks & 1;
    if (ks + 1 < nk) {
      stage(buf ^ 1, (ks + 1) << 5);
      // 8 outstanding (4 old + 4 new); in-order completion -> old tile done
      asm volatile("s_wait_asynccnt 0x4" ::: "memory");
    } else {
      asm volatile("s_wait_asynccnt 0x0" ::: "memory");
    }
    __syncthreads();  // all waves' async stores to this buffer visible

    v16bf af[2], bfm[4];
#pragma unroll
    for (int mt = 0; mt < 2; ++mt) {
      int rb = (waveM * 32 + mt * 16 + lr) * 40;
      af[mt] = combine8(*(const v8bf*)&As[buf][rb + g0],
                        *(const v8bf*)&As[buf][rb + 16 + g0]);
    }
#pragma unroll
    for (int nt = 0; nt < 4; ++nt) {
      int rb = (waveN * 64 + nt * 16 + lr) * 40;
      bfm[nt] = combine8(*(const v8bf*)&Bs[buf][rb + g0],
                         *(const v8bf*)&Bs[buf][rb + 16 + g0]);
    }
#pragma unroll
    for (int mt = 0; mt < 2; ++mt)
#pragma unroll
      for (int nt = 0; nt < 4; ++nt)
        acc[mt][nt] = wmma_bf16(af[mt], bfm[nt], acc[mt][nt]);
    __syncthreads();  // protect buffer refilled next iteration
  }

  // ---- epilogue: C/D layout -> row = VGPR idx (+8 hi lanes), col = lane%16
#pragma unroll
  for (int mt = 0; mt < 2; ++mt)
#pragma unroll
    for (int nt = 0; nt < 4; ++nt)
#pragma unroll
      for (int r = 0; r < 8; ++r) {
        int row = m0 + waveM * 32 + mt * 16 + ((lane < 16) ? r : 8 + r);
        int col = n0 + waveN * 64 + nt * 16 + lr;
        size_t o = (size_t)row * Nout + col;
        float v = acc[mt][nt][r];
        if (EPI == 2) {
          outF[o] = v + res[o];
        } else {
          outB[o] = (__bf16)v;
        }
      }
}

// ---------------------------------------------------------------------------
// Build V^T [B, H, HD, N] bf16 from qkv [B*N, 3*D] bf16 (V at column 2048)
// ---------------------------------------------------------------------------
__global__ __launch_bounds__(256) void reorg_vt_kernel(
    const __bf16* __restrict__ qkv, __bf16* __restrict__ vt) {
  size_t idx = (size_t)blockIdx.x * 256 + threadIdx.x;  // B*H*HD*N = 4194304
  int n = idx & (NN - 1);
  int d = (idx >> 11) & (HD - 1);
  int h = (idx >> 17) & (HH - 1);
  int b = (int)(idx >> 21);
  vt[idx] = qkv[(size_t)(b * NN + n) * (3 * DD) + 2 * DD + h * HD + d];
}

// ---------------------------------------------------------------------------
// Flash attention: per wave, 16 q-rows; kv chunks of 32; online softmax in
// exp2 domain using raw v_exp_f32 (args <= 0, flush-to-zero is fine).
// Row max: xor1/xor2 via DPP quad_perm (VALU), xor4/xor8 via ds_bpermute.
// Row-sums via an extra WMMA against an all-ones B matrix (C-layout rows
// match the l[] register mapping). P relayout C->A via per-wave LDS tile
// + s_wait_dscnt. grid = (N/64, B*H), block = 128 (4 waves)
// ---------------------------------------------------------------------------
__global__ __launch_bounds__(128) void flash_attn_kernel(
    const __bf16* __restrict__ qkv, const __bf16* __restrict__ vt,
    __bf16* __restrict__ o) {
  __shared__ __bf16 Ps[4][16][32];
  const int tid = threadIdx.x;
  const int lane = tid & 31, wid = tid >> 5;
  const int bh = blockIdx.y, b = bh >> 4, h = bh & 15;
  const int q0 = blockIdx.x * 64 + wid * 16;
  const int lr = lane & 15, g0 = (lane >> 4) * 8;
  const float SC = 0.125f * 1.44269504088896340736f;  // scale * log2(e)

  // Q fragments (16x64 -> two 16x32 K-steps)
  const size_t qrow = (size_t)(b * NN + q0 + lr) * (3 * DD) + h * HD;
  v16bf qa[2];
  qa[0] = combine8(*(const v8bf*)&qkv[qrow + g0],
                   *(const v8bf*)&qkv[qrow + 16 + g0]);
  qa[1] = combine8(*(const v8bf*)&qkv[qrow + 32 + g0],
                   *(const v8bf*)&qkv[qrow + 48 + g0]);

  v16bf ones;
#pragma unroll
  for (int i = 0; i < 16; ++i) ones[i] = (__bf16)1.0f;

  v8f accO[4] = {};
  v8f accL = {};
  float m[8];
#pragma unroll
  for (int r = 0; r < 8; ++r) m[r] = -__builtin_inff();

  for (int j = 0; j < NN; j += 32) {
    // ---- S = Q @ K^T (16x32), in exp2 domain ----
    v8f s[2];
#pragma unroll
    for (int nt = 0; nt < 2; ++nt) {
      const size_t krow =
          (size_t)(b * NN + j + nt * 16 + lr) * (3 * DD) + DD + h * HD;
      v16bf kb0 = combine8(*(const v8bf*)&qkv[krow + g0],
                           *(const v8bf*)&qkv[krow + 16 + g0]);
      v16bf kb1 = combine8(*(const v8bf*)&qkv[krow + 32 + g0],
                           *(const v8bf*)&qkv[krow + 48 + g0]);
      v8f a = {};
      a = wmma_bf16(qa[0], kb0, a);
      a = wmma_bf16(qa[1], kb1, a);
      s[nt] = a * SC;
    }

    // ---- row max (rows live per accumulator register) ----
    float mx[8], alpha[8];
#pragma unroll
    for (int r = 0; r < 8; ++r) mx[r] = fmaxf(s[0][r], s[1][r]);
#pragma unroll
    for (int r = 0; r < 8; ++r) {
      mx[r] = fmaxf(mx[r], dpp_qperm_f<0xB1>(mx[r]));  // xor 1 (quad_perm 1,0,3,2)
      mx[r] = fmaxf(mx[r], dpp_qperm_f<0x4E>(mx[r]));  // xor 2 (quad_perm 2,3,0,1)
    }
#pragma unroll
    for (int off = 4; off <= 8; off <<= 1)
#pragma unroll
      for (int r = 0; r < 8; ++r) mx[r] = fmaxf(mx[r], __shfl_xor(mx[r], off, 32));
#pragma unroll
    for (int r = 0; r < 8; ++r) {
      float mn = fmaxf(m[r], mx[r]);
      alpha[r] = __builtin_amdgcn_exp2f(m[r] - mn);   // raw v_exp_f32
      m[r] = mn;
    }
#pragma unroll
    for (int nt = 0; nt < 2; ++nt)
#pragma unroll
      for (int r = 0; r < 8; ++r)
        s[nt][r] = __builtin_amdgcn_exp2f(s[nt][r] - m[r]);

    // rescale running O and l
#pragma unroll
    for (int ft = 0; ft < 4; ++ft)
#pragma unroll
      for (int r = 0; r < 8; ++r) accO[ft][r] *= alpha[r];
#pragma unroll
    for (int r = 0; r < 8; ++r) accL[r] *= alpha[r];

    // ---- P: C-layout -> A-layout through per-wave LDS tile ----
    const int rlo = (lane < 16) ? 0 : 8;
#pragma unroll
    for (int nt = 0; nt < 2; ++nt)
#pragma unroll
      for (int r = 0; r < 8; ++r)
        Ps[wid][rlo + r][nt * 16 + lr] = (__bf16)s[nt][r];
    asm volatile("s_wait_dscnt 0x0" ::: "memory");  // wave-private LDS RAW
    v16bf pa = combine8(*(const v8bf*)&Ps[wid][lr][g0],
                        *(const v8bf*)&Ps[wid][lr][16 + g0]);

    // l += rowsum(P) via WMMA against ones (every column = rowsum)
    accL = wmma_bf16(pa, ones, accL);

    // ---- O += P(16x32) @ V(32x64), V^T gives contiguous key-axis loads ----
#pragma unroll
    for (int ft = 0; ft < 4; ++ft) {
      const size_t vrow = (size_t)(bh * HD + ft * 16 + lr) * NN + j;
      v16bf vb = combine8(*(const v8bf*)&vt[vrow + g0],
                          *(const v8bf*)&vt[vrow + 16 + g0]);
      accO[ft] = wmma_bf16(pa, vb, accO[ft]);
    }
  }

  // ---- finalize: O /= l, store token-major bf16 [B*N, H*HD] ----
#pragma unroll
  for (int ft = 0; ft < 4; ++ft)
#pragma unroll
    for (int r = 0; r < 8; ++r) {
      int row = q0 + ((lane < 16) ? r : 8 + r);
      int col = h * HD + ft * 16 + lr;
      o[(size_t)(b * NN + row) * DD + col] = (__bf16)(accO[ft][r] / accL[r]);
    }
}

// ---------------------------------------------------------------------------
// SwiGLU: U[m, j] = silu(G[m, j]) * G[m, MLP + j]
// ---------------------------------------------------------------------------
__global__ __launch_bounds__(256) void swiglu_kernel(
    const __bf16* __restrict__ G, __bf16* __restrict__ U) {
  size_t idx = (size_t)blockIdx.x * 256 + threadIdx.x;  // MTOK*MLP
  size_t mrow = idx >> 12;
  int j = (int)(idx & (MLP_ - 1));
  float gt = (float)G[mrow * (2 * MLP_) + j];
  float up = (float)G[mrow * (2 * MLP_) + MLP_ + j];
  // silu(g) = g / (1 + exp(-g)); exp(-g) = exp2(-g*log2e), raw v_exp_f32
  float e = __builtin_amdgcn_exp2f(-gt * 1.44269504088896340736f);
  U[idx] = (__bf16)(gt / (1.f + e) * up);
}

// ---------------------------------------------------------------------------
extern "C" void kernel_launch(void* const* d_in, const int* in_sizes, int n_in,
                              void* d_out, int out_size, void* d_ws,
                              size_t ws_size, hipStream_t stream) {
  const float* x      = (const float*)d_in[0];
  // d_in[1] = t, d_in[2] = dt : unused by the reference
  const float* w_qkv  = (const float*)d_in[3];
  const float* w_o    = (const float*)d_in[4];
  const float* w1     = (const float*)d_in[5];
  const float* w2     = (const float*)d_in[6];
  const float* g_attn = (const float*)d_in[7];
  const float* g_ff   = (const float*)d_in[8];
  float* out = (float*)d_out;
  char* ws = (char*)d_ws;

  // workspace layout (with reuse): ~152 MB total
  const size_t o_wqkv = 0;
  const size_t o_wo   = o_wqkv + (size_t)3 * DD * DD * 2;        //  6 MB
  const size_t o_w1   = o_wo   + (size_t)DD * DD * 2;            //  2 MB
  const size_t o_w2   = o_w1   + (size_t)2 * MLP_ * DD * 2;      // 16 MB
  const size_t o_h    = o_w2   + (size_t)DD * MLP_ * 2;          //  8 MB
  const size_t o_qkv  = o_h    + (size_t)MTOK * DD * 2;          //  8 MB (O reuses)
  const size_t o_vt   = o_qkv  + (size_t)MTOK * 3 * DD * 2;      // 24 MB (hf/U reuse)
  const size_t o_x2   = o_vt   + (size_t)BB * HH * HD * NN * 2;  //  8 MB
  const size_t o_G    = o_x2   + (size_t)MTOK * DD * 4;          // 16 MB
                                                                 // G: 64 MB
  __bf16* wqkvb = (__bf16*)(ws + o_wqkv);
  __bf16* wob   = (__bf16*)(ws + o_wo);
  __bf16* w1b   = (__bf16*)(ws + o_w1);
  __bf16* w2b   = (__bf16*)(ws + o_w2);
  __bf16* hb    = (__bf16*)(ws + o_h);
  __bf16* Ob    = (__bf16*)(ws + o_h);    // reuse: h dead after QKV GEMM
  __bf16* qkvb  = (__bf16*)(ws + o_qkv);
  __bf16* hfb   = (__bf16*)(ws + o_qkv);  // reuse: qkv dead after attention
  __bf16* Ub    = (__bf16*)(ws + o_qkv);  // reuse: hf+vt dead after FF1 (32 MB)
  __bf16* vtb   = (__bf16*)(ws + o_vt);
  float*  x2    = (float*)(ws + o_x2);
  __bf16* Gb    = (__bf16*)(ws + o_G);

  // 1) weights -> bf16
  cast_bf16_kernel<<<(3 * DD * DD) / 256, 256, 0, stream>>>(w_qkv, wqkvb, 3 * DD * DD);
  cast_bf16_kernel<<<(DD * DD) / 256, 256, 0, stream>>>(w_o, wob, DD * DD);
  cast_bf16_kernel<<<(2 * MLP_ * DD) / 256, 256, 0, stream>>>(w1, w1b, 2 * MLP_ * DD);
  cast_bf16_kernel<<<(DD * MLP_) / 256, 256, 0, stream>>>(w2, w2b, DD * MLP_);

  // 2) h = RMSNorm(x) * g_attn  (bf16)
  rmsnorm_kernel<<<MTOK, 256, 0, stream>>>(x, g_attn, hb);

  // 3) qkv = h @ w_qkv^T  [4096, 3072] bf16
  gemm_bf16_kernel<1><<<dim3((3 * DD) / 128, MTOK / 128), 256, 0, stream>>>(
      hb, wqkvb, DD, 3 * DD, nullptr, nullptr, qkvb);

  // 4) V^T [B,H,HD,N]
  reorg_vt_kernel<<<(BB * HH * HD * NN) / 256, 256, 0, stream>>>(qkvb, vtb);

  // 5) flash attention -> O [4096, 1024] bf16
  flash_attn_kernel<<<dim3(NN / 64, BB * HH), 128, 0, stream>>>(qkvb, vtb, Ob);

  // 6) x2 = x + O @ w_o^T  (f32)
  gemm_bf16_kernel<2><<<dim3(DD / 128, MTOK / 128), 256, 0, stream>>>(
      Ob, wob, DD, DD, x, x2, nullptr);

  // 7) hf = RMSNorm(x2) * g_ff  (bf16)
  rmsnorm_kernel<<<MTOK, 256, 0, stream>>>(x2, g_ff, hfb);

  // 8) G = hf @ w1^T  [4096, 8192] bf16
  gemm_bf16_kernel<1><<<dim3((2 * MLP_) / 128, MTOK / 128), 256, 0, stream>>>(
      hfb, w1b, DD, 2 * MLP_, nullptr, nullptr, Gb);

  // 9) U = silu(gate) * up  [4096, 4096] bf16
  swiglu_kernel<<<(size_t)(MTOK) * MLP_ / 256, 256, 0, stream>>>(Gb, Ub);

  // 10) out = x2 + U @ w2^T  (f32)
  gemm_bf16_kernel<2><<<dim3(DD / 128, MTOK / 128), 256, 0, stream>>>(
      Ub, w2b, MLP_, DD, x2, out, nullptr);
}
